// CrossAttentionFusion_47227460387468
// MI455X (gfx1250) — matrix operand order repaired and checked by reading, not compiled, not run on hardware
//
#include <hip/hip_runtime.h>
#include <hip/hip_bf16.h>

// Problem constants (match reference)
#define S2C   256
#define DEMC  64
#define OUTC  256
#define BATCH 4
#define NPIX  4096          // 64*64
#define NT    (NPIX / 16)   // 256 j/i tiles per batch
#define SCALE 0.0625f       // 256^-0.5

typedef __attribute__((ext_vector_type(16))) __bf16       v16bf;
typedef __attribute__((ext_vector_type(8)))  float        v8f;
typedef __attribute__((ext_vector_type(4)))  unsigned int u32x4;
typedef __attribute__((ext_vector_type(4)))  float        f32x4;

union UBF {
    v16bf v;
    u32x4 q[2];
    unsigned int w[8];
};

__device__ __forceinline__ unsigned short f2bf(float f) {
    unsigned int b = __float_as_uint(f);
    b += 0x7FFFu + ((b >> 16) & 1u);   // round-to-nearest-even
    return (unsigned short)(b >> 16);
}

// B-operand tile: lane = col j (lane&15); half hi reads chunks {c0+hi*8, c0+16+hi*8}
__device__ __forceinline__ void load_btile(UBF (&B)[8], const unsigned short* qbase, int j) {
#pragma unroll
    for (int cc = 0; cc < 8; ++cc) {
        const unsigned short* p = qbase + j * OUTC + cc * 32;
        B[cc].q[0] = *reinterpret_cast<const u32x4*>(p);
        B[cc].q[1] = *reinterpret_cast<const u32x4*>(p + 16);
    }
}

// A-operand tile: lane = row (lane&15); half hi reads chunks {c0+hi*8, c0+16+hi*8}
__device__ __forceinline__ void load_atile(UBF (&A)[8], const unsigned short* row) {
#pragma unroll
    for (int cc = 0; cc < 8; ++cc) {
        const unsigned short* p = row + cc * 32;
        A[cc].q[0] = *reinterpret_cast<const u32x4*>(p);
        A[cc].q[1] = *reinterpret_cast<const u32x4*>(p + 16);
    }
}

// 16x16 score tile over the full 256-channel contraction (8 WMMAs)
__device__ __forceinline__ v8f score_tile(const UBF (&A)[8], const UBF (&B)[8]) {
    v8f S = {0.f, 0.f, 0.f, 0.f, 0.f, 0.f, 0.f, 0.f};
#pragma unroll
    for (int cc = 0; cc < 8; ++cc)
        S = __builtin_amdgcn_wmma_f32_16x16x32_bf16(false, A[cc].v, false, B[cc].v,
                                                    (short)0, S, false, false);
    return S;
}

// Per-lane online softmax stats update (no cross-lane traffic in the hot loop)
__device__ __forceinline__ void stats_update(v8f S, float (&mr)[8], float (&zr)[8]) {
#pragma unroll
    for (int r = 0; r < 8; ++r) {
        float s  = S[r] * SCALE;
        float nm = fmaxf(mr[r], s);
        zr[r] = zr[r] * __expf(mr[r] - nm) + __expf(s - nm);
        mr[r] = nm;
    }
}

// ---------------------------------------------------------------------------
// Kernel 1: 1x1-conv projections -> bf16 Q/K/V in WMMA-friendly layouts.
//   Qt[b][n][c], Kt[b][n][c]  (pixel-major rows of 256 channels)
//   Vc[b][c][n]               (channel-major rows of 4096 pixels)
// ---------------------------------------------------------------------------
__global__ __launch_bounds__(256) void qkv_proj_kernel(
    const float* __restrict__ xs2, const float* __restrict__ xdem,
    const float* __restrict__ Wq, const float* __restrict__ bq,
    const float* __restrict__ Wk, const float* __restrict__ bk,
    const float* __restrict__ Wv, const float* __restrict__ bv,
    unsigned short* __restrict__ Qt, unsigned short* __restrict__ Kt,
    unsigned short* __restrict__ Vc) {
    const int b  = blockIdx.x >> 8;
    const int n0 = (blockIdx.x & 255) << 4;
    const int o  = threadIdx.x;

    __shared__ float lxs[S2C][16];
    __shared__ float lxd[DEMC][16];

    for (int l = threadIdx.x; l < S2C * 16; l += 256) {
        int c = l >> 4, nn = l & 15;
        lxs[c][nn] = xs2[(b * S2C + c) * NPIX + n0 + nn];
    }
    for (int l = threadIdx.x; l < DEMC * 16; l += 256) {
        int c = l >> 4, nn = l & 15;
        lxd[c][nn] = xdem[(b * DEMC + c) * NPIX + n0 + nn];
    }
    __syncthreads();

    // ---- Q = Wq * xs2 + bq ----
    float accq[16];
#pragma unroll
    for (int i = 0; i < 16; ++i) accq[i] = 0.f;
#pragma unroll 2
    for (int c = 0; c < S2C; c += 4) {
        f32x4 wr = *reinterpret_cast<const f32x4*>(&Wq[o * S2C + c]);
#pragma unroll
        for (int cs = 0; cs < 4; ++cs) {
            const f32x4* xr = reinterpret_cast<const f32x4*>(&lxs[c + cs][0]);
#pragma unroll
            for (int k = 0; k < 4; ++k) {
                f32x4 xv = xr[k];
#pragma unroll
                for (int e = 0; e < 4; ++e)
                    accq[k * 4 + e] = fmaf(wr[cs], xv[e], accq[k * 4 + e]);
            }
        }
    }
    float bqo = bq[o];
#pragma unroll
    for (int nn = 0; nn < 16; ++nn)
        Qt[(b * NPIX + n0 + nn) * OUTC + o] = f2bf(accq[nn] + bqo);

    // ---- K, V = Wk/Wv * xdem + bk/bv ----
    float acck[16], accv[16];
#pragma unroll
    for (int i = 0; i < 16; ++i) { acck[i] = 0.f; accv[i] = 0.f; }
#pragma unroll 2
    for (int c = 0; c < DEMC; c += 4) {
        f32x4 wkr = *reinterpret_cast<const f32x4*>(&Wk[o * DEMC + c]);
        f32x4 wvr = *reinterpret_cast<const f32x4*>(&Wv[o * DEMC + c]);
#pragma unroll
        for (int cs = 0; cs < 4; ++cs) {
            const f32x4* xr = reinterpret_cast<const f32x4*>(&lxd[c + cs][0]);
#pragma unroll
            for (int k = 0; k < 4; ++k) {
                f32x4 xv = xr[k];
#pragma unroll
                for (int e = 0; e < 4; ++e) {
                    acck[k * 4 + e] = fmaf(wkr[cs], xv[e], acck[k * 4 + e]);
                    accv[k * 4 + e] = fmaf(wvr[cs], xv[e], accv[k * 4 + e]);
                }
            }
        }
    }
    float bko = bk[o], bvo = bv[o];
#pragma unroll
    for (int nn = 0; nn < 16; ++nn)
        Kt[(b * NPIX + n0 + nn) * OUTC + o] = f2bf(acck[nn] + bko);

    unsigned int vp[8];
#pragma unroll
    for (int p = 0; p < 8; ++p)
        vp[p] = (unsigned int)f2bf(accv[2 * p] + bvo) |
                ((unsigned int)f2bf(accv[2 * p + 1] + bvo) << 16);
    u32x4* vdst = reinterpret_cast<u32x4*>(&Vc[(b * OUTC + o) * NPIX + n0]);
    vdst[0] = (u32x4){vp[0], vp[1], vp[2], vp[3]};
    vdst[1] = (u32x4){vp[4], vp[5], vp[6], vp[7]};
}

// ---------------------------------------------------------------------------
// Kernel 2: logsumexp row statistics. S = K^T Q via WMMA bf16.
// Per-lane online (m, Z) over the j-loop (lane sees columns j%16==lane%16);
// single cross-lane merge at the end. B tiles double-buffered so the next
// tile's 16 b128 loads overlap the current tile's 8 WMMAs.
// Output: Lrow[i] = max_j(s) + log(sum_j exp(s - max)).
// ---------------------------------------------------------------------------
__global__ __launch_bounds__(128) void attn_stats_kernel(
    const unsigned short* __restrict__ Qt, const unsigned short* __restrict__ Kt,
    float* __restrict__ Lrow) {
    const int lane = threadIdx.x & 31;
    const int wv   = threadIdx.x >> 5;
    const int lp   = lane & 15;
    const int hi   = lane >> 4;
    const int idx  = blockIdx.x * 4 + wv;
    const int b    = idx >> 8;
    const int i0   = (idx & 255) << 4;

    UBF A[8];
    load_atile(A, Kt + (b * NPIX + i0 + lp) * OUTC + hi * 8);

    float mr[8], zr[8];
#pragma unroll
    for (int r = 0; r < 8; ++r) { mr[r] = -1e30f; zr[r] = 0.f; }

    const unsigned short* qbase = Qt + (b * NPIX + lp) * OUTC + hi * 8;

    UBF B0[8], B1[8];
    load_btile(B0, qbase, 0);
    for (int j0 = 0; j0 < NPIX; j0 += 32) {
        load_btile(B1, qbase, j0 + 16);              // prefetch while WMMAs run
        stats_update(score_tile(A, B0), mr, zr);
        if (j0 + 32 < NPIX) load_btile(B0, qbase, j0 + 32);
        stats_update(score_tile(A, B1), mr, zr);
    }

    // merge the 16 per-lane partials of each row (stays within 16-lane half)
#pragma unroll
    for (int r = 0; r < 8; ++r) {
        float m = mr[r], z = zr[r];
#pragma unroll
        for (int msk = 1; msk <= 8; msk <<= 1) {
            float om = __shfl_xor(m, msk, 32);
            float oz = __shfl_xor(z, msk, 32);
            float nm = fmaxf(m, om);
            z = z * __expf(m - nm) + oz * __expf(om - nm);
            m = nm;
        }
        mr[r] = m; zr[r] = z;
    }
    if (lp == 0) {
#pragma unroll
        for (int r = 0; r < 8; ++r)
            Lrow[b * NPIX + i0 + hi * 8 + r] = mr[r] + __logf(zr[r]);
    }
}

// ---------------------------------------------------------------------------
// Kernel 3: Out[:, j-tile] = V * softmax(K^T Q)[:, j-tile] + residual.
// Score tiles recomputed with WMMA (Q strip resident in VGPRs), normalized
// via exp(s - logsumexp), transposed through LDS, then 16 c-tile WMMA
// accumulations (V loads batched 4 tiles at a time to clause the b128s).
// ---------------------------------------------------------------------------
__global__ __launch_bounds__(128, 1) void attn_out_kernel(
    const unsigned short* __restrict__ Qt, const unsigned short* __restrict__ Kt,
    const unsigned short* __restrict__ Vc, const float* __restrict__ Lrow,
    const float* __restrict__ xs2, float* __restrict__ out) {
    const int lane = threadIdx.x & 31;
    const int wv   = threadIdx.x >> 5;
    const int lp   = lane & 15;
    const int hi   = lane >> 4;
    const int idx  = blockIdx.x * 4 + wv;
    const int b    = idx >> 8;
    const int j0   = (idx & 255) << 4;

    // [wave][j(16)][i(32) + pad to 40 halves] -- padded rows kill bank conflicts
    __shared__ __align__(16) unsigned short ptile[4][16][40];

    v8f acc[16];
    v8f zero = {0.f, 0.f, 0.f, 0.f, 0.f, 0.f, 0.f, 0.f};
#pragma unroll
    for (int t = 0; t < 16; ++t) acc[t] = zero;

    // Q j-tile strip (B operand) reused for all 256 i-tiles: keep in VGPRs.
    UBF Bq[8];
    load_btile(Bq, Qt + (b * NPIX + lp) * OUTC + hi * 8, j0);

    for (int i0 = 0; i0 < NPIX; i0 += 32) {
        if (i0 + 32 < NPIX)   // warm L2/WGP$ for the next K strip
            __builtin_prefetch(Kt + (b * NPIX + i0 + 32 + lp) * OUTC, 0, 1);

#pragma unroll
        for (int sub = 0; sub < 2; ++sub) {
            const int ib = i0 + sub * 16;
            // logsumexp for rows ib + 8*hi + 0..7 (hoisted so load overlaps WMMA)
            const float* lb = Lrow + b * NPIX + ib + hi * 8;
            f32x4 l0 = *reinterpret_cast<const f32x4*>(lb);
            f32x4 l1 = *reinterpret_cast<const f32x4*>(lb + 4);

            UBF Ak[8];
            load_atile(Ak, Kt + (b * NPIX + ib + lp) * OUTC + hi * 8);
            v8f S = score_tile(Ak, Bq);

            float lv[8];
#pragma unroll
            for (int r = 0; r < 4; ++r) { lv[r] = l0[r]; lv[r + 4] = l1[r]; }
            unsigned int pk[4];
#pragma unroll
            for (int rp = 0; rp < 4; ++rp) {
                float plo = __expf(S[2 * rp]     * SCALE - lv[2 * rp]);
                float phi = __expf(S[2 * rp + 1] * SCALE - lv[2 * rp + 1]);
                pk[rp] = (unsigned int)f2bf(plo) | ((unsigned int)f2bf(phi) << 16);
            }
            // D-layout rows r+8*hi are contiguous per lane -> transposed LDS
            // write lands as one 16B chunk at [j=lp][i_rel = sub*16 + hi*8].
            *reinterpret_cast<u32x4*>(&ptile[wv][lp][sub * 16 + hi * 8]) =
                (u32x4){pk[0], pk[1], pk[2], pk[3]};
        }

        // Re-read P tile in B-operand layout (lane = col j, K-run over i).
        // Same-wave LDS ops are in-order: no barrier needed.
        const unsigned short* lrd = &ptile[wv][lp][hi * 8];
        UBF Bp;
        Bp.q[0] = *reinterpret_cast<const u32x4*>(lrd);
        Bp.q[1] = *reinterpret_cast<const u32x4*>(lrd + 16);

        // Out accumulation: batch V-tile loads 4 c-tiles at a time so the 8
        // b128 loads clause together and overlap the 4 WMMAs.
#pragma unroll
        for (int cg = 0; cg < 4; ++cg) {
            UBF Av[4];
#pragma unroll
            for (int t = 0; t < 4; ++t) {
                const unsigned short* vp =
                    Vc + (b * OUTC + (cg * 4 + t) * 16 + lp) * NPIX + i0 + hi * 8;
                Av[t].q[0] = *reinterpret_cast<const u32x4*>(vp);
                Av[t].q[1] = *reinterpret_cast<const u32x4*>(vp + 16);
            }
#pragma unroll
            for (int t = 0; t < 4; ++t)
                acc[cg * 4 + t] = __builtin_amdgcn_wmma_f32_16x16x32_bf16(
                    false, Av[t].v, false, Bp.v, (short)0, acc[cg * 4 + t], false, false);
        }
    }

    // Epilogue: residual add + store f32. D element (c = ct*16 + r + 8*hi, j = j0+lp)
#pragma unroll
    for (int ct = 0; ct < 16; ++ct) {
#pragma unroll
        for (int r = 0; r < 8; ++r) {
            int c    = ct * 16 + r + 8 * hi;
            int oidx = (b * OUTC + c) * NPIX + j0 + lp;
            out[oidx] = acc[ct][r] + xs2[oidx];
        }
    }
}

// ---------------------------------------------------------------------------
extern "C" void kernel_launch(void* const* d_in, const int* in_sizes, int n_in,
                              void* d_out, int out_size, void* d_ws, size_t ws_size,
                              hipStream_t stream) {
    const float* xs2  = (const float*)d_in[0];
    const float* xdem = (const float*)d_in[1];
    const float* Wq   = (const float*)d_in[2];
    const float* bq   = (const float*)d_in[3];
    const float* Wk   = (const float*)d_in[4];
    const float* bk   = (const float*)d_in[5];
    const float* Wv   = (const float*)d_in[6];
    const float* bv   = (const float*)d_in[7];
    float* out = (float*)d_out;

    // Workspace layout: Qt | Kt | Vc (bf16, 8 MB each) | Lrow (f32, 64 KB)
    char* ws = (char*)d_ws;
    const size_t szQ = (size_t)BATCH * NPIX * OUTC * sizeof(unsigned short);
    unsigned short* Qt = (unsigned short*)ws;
    unsigned short* Kt = (unsigned short*)(ws + szQ);
    unsigned short* Vc = (unsigned short*)(ws + 2 * szQ);
    float* Lrow = (float*)(ws + 3 * szQ);

    qkv_proj_kernel<<<dim3(BATCH * NT), dim3(256), 0, stream>>>(
        xs2, xdem, Wq, bq, Wk, bk, Wv, bv, Qt, Kt, Vc);
    attn_stats_kernel<<<dim3(BATCH * NT / 4), dim3(128), 0, stream>>>(
        Qt, Kt, Lrow);
    attn_out_kernel<<<dim3(BATCH * NT / 4), dim3(128), 0, stream>>>(
        Qt, Kt, Vc, Lrow, xs2, out);
}
